// GraniteMoeAttention_47536698032451
// MI455X (gfx1250) — matrix operand order repaired and checked by reading, not compiled, not run on hardware
//
#include <hip/hip_runtime.h>
#include <hip/hip_bf16.h>

typedef __attribute__((ext_vector_type(16))) __bf16 v16bf;
typedef __attribute__((ext_vector_type(8)))  float  v8f;
typedef __attribute__((ext_vector_type(2)))  __bf16 bf16x2;
typedef __attribute__((__vector_size__(16))) int    i32x4_;

#define B_    2
#define S_    2048
#define HID_  4096
#define NH_   32
#define NKV_  8
#define HD_   128
#define HALF_ 64
#define QSZ_  4096
#define KVSZ_ 1024
#define SCALE_ (1.0f / 128.0f)
#define NEG_INF_ (-3.0e38f)

#if __has_builtin(__builtin_amdgcn_global_load_async_to_lds_b128)
#define USE_ASYNC_LDS 1
#else
#define USE_ASYNC_LDS 0
#endif

__device__ __forceinline__ void wait_async0() {
#if __has_builtin(__builtin_amdgcn_s_wait_asynccnt)
  __builtin_amdgcn_s_wait_asynccnt(0);
#else
  asm volatile("s_wait_asynccnt 0" ::: "memory");
#endif
}

// ---------------------------------------------------------------------------
// WMMA helpers (CDNA5 gfx1250, wave32)
// ---------------------------------------------------------------------------
__device__ __forceinline__ v8f wmma_bf16(v16bf a, v16bf b, v8f c) {
  // (neg_a, A, neg_b, B, c_mod, C, reuse_a, reuse_b)
  return __builtin_amdgcn_wmma_f32_16x16x32_bf16(false, a, false, b, (short)0, c,
                                                 false, false);
}

__device__ __forceinline__ v8f v8f_zero() {
  v8f z;
#pragma unroll
  for (int i = 0; i < 8; ++i) z[i] = 0.0f;
  return z;
}

// A-matrix 16x32 bf16 fragment from row-major tile.
// Lane holds row M = lane%16; VGPR i holds K pair at 2i + (i>=4?8:0) + (lane>=16?8:0).
__device__ __forceinline__ v16bf load_a_frag(const __bf16* t, int pitch, int row,
                                             int kbase, int lane) {
  v16bf f;
  const int hk = (lane >= 16) ? 8 : 0;
#pragma unroll
  for (int i = 0; i < 8; ++i) {
    int k = kbase + 2 * i + ((i >= 4) ? 8 : 0) + hk;
    bf16x2 p = *(const bf16x2*)(t + (size_t)row * pitch + k);
    f[2 * i] = p.x;
    f[2 * i + 1] = p.y;
  }
  return f;
}

// B-matrix 32x16 bf16 fragment where B(k,n) = tile[n][k] (tile row-major over n).
// Lane holds col N = lane%16; VGPR i holds K pair at 2i + (i>=4?8:0) + (lane>=16?8:0).
__device__ __forceinline__ v16bf load_b_frag(const __bf16* t, int pitch, int nrow,
                                             int kbase, int lane) {
  v16bf f;
  const int hk = (lane >= 16) ? 8 : 0;
#pragma unroll
  for (int i = 0; i < 8; ++i) {
    int k = kbase + 2 * i + ((i >= 4) ? 8 : 0) + hk;
    bf16x2 p = *(const bf16x2*)(t + (size_t)nrow * pitch + k);
    f[2 * i] = p.x;
    f[2 * i + 1] = p.y;
  }
  return f;
}

// ---------------------------------------------------------------------------
// Kernel 1: qkv = hidden @ w_qkv^T, scatter to bf16 q/k/v buffers.
// M = B*S = 4096, N = 6144, K = 4096. Grid (48, 32), 256 threads (8 waves).
// ---------------------------------------------------------------------------
__global__ __launch_bounds__(256) void qkv_gemm_kernel(
    const float* __restrict__ hid, const float* __restrict__ wqkv,
    __bf16* __restrict__ qb, __bf16* __restrict__ kb, __bf16* __restrict__ vb) {
  __shared__ __bf16 At[128][34];
  __shared__ __bf16 Bt[128][34];
  const int tid = threadIdx.x, lane = tid & 31, wid = tid >> 5;
  const int bm = blockIdx.y, bn = blockIdx.x;
  const int wrow = wid >> 2, wcol = wid & 3;  // 2 x 4 wave grid
  const int l15 = lane & 15;

  v8f acc[4][2];
#pragma unroll
  for (int m = 0; m < 4; ++m)
#pragma unroll
    for (int n = 0; n < 2; ++n) acc[m][n] = v8f_zero();

  for (int kk = 0; kk < HID_; kk += 32) {
    __syncthreads();
#pragma unroll
    for (int j = 0; j < 16; ++j) {
      int idx = j * 256 + tid;
      int r = idx >> 5, c = idx & 31;
      At[r][c] = (__bf16)hid[(size_t)(bm * 128 + r) * HID_ + kk + c];
      Bt[r][c] = (__bf16)wqkv[(size_t)(bn * 128 + r) * HID_ + kk + c];
    }
    __syncthreads();
    if (kk + 32 < HID_) {  // prefetch next k-tile (L2-resident weights)
      int r = tid >> 5, c = tid & 31;
      __builtin_prefetch(&hid[(size_t)(bm * 128 + r) * HID_ + kk + 32 + c], 0, 0);
      __builtin_prefetch(&wqkv[(size_t)(bn * 128 + r) * HID_ + kk + 32 + c], 0, 0);
    }
    v16bf af[4], bfr[2];
#pragma unroll
    for (int m = 0; m < 4; ++m)
      af[m] = load_a_frag(&At[0][0], 34, wrow * 64 + m * 16 + l15, 0, lane);
#pragma unroll
    for (int n = 0; n < 2; ++n)
      bfr[n] = load_b_frag(&Bt[0][0], 34, wcol * 32 + n * 16 + l15, 0, lane);
#pragma unroll
    for (int m = 0; m < 4; ++m)
#pragma unroll
      for (int n = 0; n < 2; ++n) acc[m][n] = wmma_bf16(af[m], bfr[n], acc[m][n]);
  }

  // Block-uniform scatter: segment boundaries (4096, 5120) are 128-aligned,
  // so a whole N-tile maps to exactly one of q/k/v.
  __bf16* dbase;
  int dstride, dcol0;
  if (bn < 32) {
    dbase = qb; dstride = QSZ_; dcol0 = bn * 128;
  } else if (bn < 40) {
    dbase = kb; dstride = KVSZ_; dcol0 = bn * 128 - QSZ_;
  } else {
    dbase = vb; dstride = KVSZ_; dcol0 = bn * 128 - QSZ_ - KVSZ_;
  }
  const int mh = (lane >= 16) ? 8 : 0;
#pragma unroll
  for (int m = 0; m < 4; ++m)
#pragma unroll
    for (int n = 0; n < 2; ++n)
#pragma unroll
      for (int r = 0; r < 8; ++r) {
        int gm = bm * 128 + wrow * 64 + m * 16 + r + mh;
        int col = dcol0 + wcol * 32 + n * 16 + l15;
        dbase[(size_t)gm * dstride + col] = (__bf16)acc[m][n][r];
      }
}

// ---------------------------------------------------------------------------
// Kernel 2: NeoX RoPE applied in place to q and k (bf16).
// ---------------------------------------------------------------------------
__global__ __launch_bounds__(256) void rope_kernel(const int* __restrict__ pos,
                                                   __bf16* __restrict__ qb,
                                                   __bf16* __restrict__ kb) {
  size_t t = (size_t)blockIdx.x * 256 + threadIdx.x;
  const size_t total = (size_t)B_ * S_ * (NH_ + NKV_) * HALF_;
  if (t >= total) return;
  int i = (int)(t & (HALF_ - 1));
  size_t hr = t >> 6;
  int head = (int)(hr % (NH_ + NKV_));
  size_t bs = hr / (NH_ + NKV_);
  float p = (float)pos[bs];

  __bf16* base = (head < NH_) ? (qb + bs * QSZ_ + (size_t)head * HD_)
                              : (kb + bs * KVSZ_ + (size_t)(head - NH_) * HD_);
  float inv = __powf(10000.0f, -(float)i * (1.0f / (float)HALF_));
  float f = p * inv;
  float sn, cs;
  __sincosf(f, &sn, &cs);
  float x1 = (float)base[i];
  float x2 = (float)base[i + HALF_];
  base[i] = (__bf16)(x1 * cs - x2 * sn);
  base[i + HALF_] = (__bf16)(x2 * cs + x1 * sn);
}

// ---------------------------------------------------------------------------
// Kernel 3: causal flash attention with GQA (rep=4), all-WMMA.
// Grid: (B*NH, S/128). 256 threads = 8 waves; wave w owns q rows [128*qt+16w, +16).
// ---------------------------------------------------------------------------
__global__ __launch_bounds__(256) void flash_attn_kernel(
    const __bf16* __restrict__ qb, const __bf16* __restrict__ kb,
    const __bf16* __restrict__ vb, __bf16* __restrict__ ab) {
  __shared__ __bf16 Kt[64][130];    // K tile, row-major [kv][d]
  __shared__ __bf16 VtT[128][66];   // V tile, transposed [d][kv]
  __shared__ __bf16 Pt[8][16][66];  // per-wave P scratch (C->A layout bounce)

  const int tid = threadIdx.x, lane = tid & 31, wid = tid >> 5;
  const int bh = blockIdx.x;
  const int b = bh >> 5, h = bh & 31;
  const int kvh = h >> 2;
  const int qt = blockIdx.y;
  const int l15 = lane & 15;
  const int mh = (lane >= 16) ? 8 : 0;

  // Q fragments (16 rows x 128 d) kept in registers for the whole kernel.
  v16bf qa[4];
  {
    int row_s = qt * 128 + wid * 16 + l15;
    const __bf16* qp = qb + (((size_t)(b * S_ + row_s)) * NH_ + h) * HD_;
    const int hk = (lane >= 16) ? 8 : 0;
#pragma unroll
    for (int f = 0; f < 4; ++f) {
#pragma unroll
      for (int i = 0; i < 8; ++i) {
        int kd = f * 32 + 2 * i + ((i >= 4) ? 8 : 0) + hk;
        bf16x2 p = *(const bf16x2*)(qp + kd);
        qa[f][2 * i] = p.x;
        qa[f][2 * i + 1] = p.y;
      }
    }
  }

  v8f o[8];
#pragma unroll
  for (int nf = 0; nf < 8; ++nf) o[nf] = v8f_zero();
  float mi[8], li[8];
#pragma unroll
  for (int r = 0; r < 8; ++r) { mi[r] = NEG_INF_; li[r] = 0.0f; }

  const int ktmax = 2 * qt + 2;  // 64-wide kv tiles covering causal span
  for (int kt = 0; kt < ktmax; ++kt) {
    __syncthreads();  // previous iteration done reading Kt/VtT/Pt

    // --- K tile: async copy global -> LDS (ASYNCcnt path) when available ---
#if USE_ASYNC_LDS
#pragma unroll
    for (int j = 0; j < 4; ++j) {
      int idx = j * 256 + tid;  // 1024 chunks of 8 bf16 (16B)
      int r = idx >> 4;
      int c = (idx & 15) * 8;
      const __bf16* src =
          kb + (((size_t)(b * S_ + kt * 64 + r)) * NKV_ + kvh) * HD_ + c;
      __builtin_amdgcn_global_load_async_to_lds_b128(
          (i32x4_*)(uintptr_t)src, (i32x4_*)(&Kt[r][c]), 0, 0);
    }
#else
#pragma unroll
    for (int j = 0; j < 32; ++j) {
      int idx = j * 256 + tid;
      int r = idx >> 7, c = idx & 127;
      Kt[r][c] = kb[(((size_t)(b * S_ + kt * 64 + r)) * NKV_ + kvh) * HD_ + c];
    }
#endif
    // --- V tile: transposed store so P@V B-fragments pair-load ---
#pragma unroll
    for (int j = 0; j < 32; ++j) {
      int idx = j * 256 + tid;
      int r = idx >> 7, c = idx & 127;
      VtT[c][r] = vb[(((size_t)(b * S_ + kt * 64 + r)) * NKV_ + kvh) * HD_ + c];
    }
#if USE_ASYNC_LDS
    wait_async0();
#endif
    __syncthreads();

    // S = Q @ K^T  (16 x 64 per wave)
    v8f sa[4];
#pragma unroll
    for (int nf = 0; nf < 4; ++nf) sa[nf] = v8f_zero();
#pragma unroll
    for (int kkd = 0; kkd < 4; ++kkd) {
#pragma unroll
      for (int nf = 0; nf < 4; ++nf) {
        v16bf bk = load_b_frag(&Kt[0][0], 130, nf * 16 + l15, kkd * 32, lane);
        sa[nf] = wmma_bf16(qa[kkd], bk, sa[nf]);
      }
    }

    // scale + causal mask (only diagonal tiles need masking) + online softmax
    const bool needMask = (kt >= 2 * qt);
    float mnew[8];
#pragma unroll
    for (int r = 0; r < 8; ++r) {
      int qrow = qt * 128 + wid * 16 + r + mh;
      float rowmax = NEG_INF_;
#pragma unroll
      for (int nf = 0; nf < 4; ++nf) {
        float sv = sa[nf][r] * SCALE_;
        if (needMask) {
          int kc = kt * 64 + nf * 16 + l15;
          sv = (kc <= qrow) ? sv : NEG_INF_;
        }
        sa[nf][r] = sv;
        rowmax = fmaxf(rowmax, sv);
      }
#pragma unroll
      for (int msk = 1; msk < 16; msk <<= 1)
        rowmax = fmaxf(rowmax, __shfl_xor(rowmax, msk, 32));
      mnew[r] = fmaxf(mi[r], rowmax);
    }
#pragma unroll
    for (int r = 0; r < 8; ++r) {
      float alpha = __expf(mi[r] - mnew[r]);
      float rsum = 0.0f;
#pragma unroll
      for (int nf = 0; nf < 4; ++nf) {
        float p = __expf(sa[nf][r] - mnew[r]);
        sa[nf][r] = p;
        rsum += p;
      }
#pragma unroll
      for (int msk = 1; msk < 16; msk <<= 1) rsum += __shfl_xor(rsum, msk, 32);
      li[r] = li[r] * alpha + rsum;
      mi[r] = mnew[r];
#pragma unroll
      for (int nf = 0; nf < 8; ++nf) o[nf][r] *= alpha;
    }

    // C-layout -> A-layout via per-wave LDS scratch
#pragma unroll
    for (int r = 0; r < 8; ++r) {
      int prow = r + mh;
#pragma unroll
      for (int nf = 0; nf < 4; ++nf)
        Pt[wid][prow][nf * 16 + l15] = (__bf16)sa[nf][r];
    }
    __syncthreads();

    // O += P @ V  (16 x 128 per wave, K = 64); V fragments pair-load from VtT
    v16bf pa[2];
#pragma unroll
    for (int kkp = 0; kkp < 2; ++kkp)
      pa[kkp] = load_a_frag(&Pt[wid][0][0], 66, l15, kkp * 32, lane);
#pragma unroll
    for (int nf = 0; nf < 8; ++nf) {
#pragma unroll
      for (int kkp = 0; kkp < 2; ++kkp) {
        v16bf vbf = load_b_frag(&VtT[0][0], 66, nf * 16 + l15, kkp * 32, lane);
        o[nf] = wmma_bf16(pa[kkp], vbf, o[nf]);
      }
    }
  }

  // epilogue: normalize and write attn (B, S, NH*HD) bf16
#pragma unroll
  for (int r = 0; r < 8; ++r) {
    float inv = 1.0f / li[r];
    int row_s = qt * 128 + wid * 16 + r + mh;
#pragma unroll
    for (int nf = 0; nf < 8; ++nf) {
      int d = nf * 16 + l15;
      ab[((size_t)(b * S_ + row_s)) * QSZ_ + h * HD_ + d] =
          (__bf16)(o[nf][r] * inv);
    }
  }
}

// ---------------------------------------------------------------------------
// Kernel 4: out = attn @ w_o^T. M = 4096, N = 4096, K = 4096. Grid (32, 32).
// ---------------------------------------------------------------------------
__global__ __launch_bounds__(256) void oproj_gemm_kernel(
    const __bf16* __restrict__ ab, const float* __restrict__ wo,
    float* __restrict__ out) {
  __shared__ __bf16 At[128][34];
  __shared__ __bf16 Bt[128][34];
  const int tid = threadIdx.x, lane = tid & 31, wid = tid >> 5;
  const int bm = blockIdx.y, bn = blockIdx.x;
  const int wrow = wid >> 2, wcol = wid & 3;
  const int l15 = lane & 15;

  v8f acc[4][2];
#pragma unroll
  for (int m = 0; m < 4; ++m)
#pragma unroll
    for (int n = 0; n < 2; ++n) acc[m][n] = v8f_zero();

  for (int kk = 0; kk < QSZ_; kk += 32) {
    __syncthreads();
#pragma unroll
    for (int j = 0; j < 16; ++j) {
      int idx = j * 256 + tid;
      int r = idx >> 5, c = idx & 31;
      At[r][c] = ab[(size_t)(bm * 128 + r) * QSZ_ + kk + c];
      Bt[r][c] = (__bf16)wo[(size_t)(bn * 128 + r) * QSZ_ + kk + c];
    }
    __syncthreads();
    if (kk + 32 < QSZ_) {
      int r = tid >> 5, c = tid & 31;
      __builtin_prefetch(&ab[(size_t)(bm * 128 + r) * QSZ_ + kk + 32 + c], 0, 0);
      __builtin_prefetch(&wo[(size_t)(bn * 128 + r) * QSZ_ + kk + 32 + c], 0, 0);
    }
    v16bf af[4], bfr[2];
#pragma unroll
    for (int m = 0; m < 4; ++m)
      af[m] = load_a_frag(&At[0][0], 34, wrow * 64 + m * 16 + l15, 0, lane);
#pragma unroll
    for (int n = 0; n < 2; ++n)
      bfr[n] = load_b_frag(&Bt[0][0], 34, wcol * 32 + n * 16 + l15, 0, lane);
#pragma unroll
    for (int m = 0; m < 4; ++m)
#pragma unroll
      for (int n = 0; n < 2; ++n) acc[m][n] = wmma_bf16(af[m], bfr[n], acc[m][n]);
  }

  const int mh = (lane >= 16) ? 8 : 0;
#pragma unroll
  for (int m = 0; m < 4; ++m)
#pragma unroll
    for (int n = 0; n < 2; ++n)
#pragma unroll
      for (int r = 0; r < 8; ++r) {
        int gm = bm * 128 + wrow * 64 + m * 16 + r + mh;
        int gn = bn * 128 + wcol * 32 + n * 16 + l15;
        out[(size_t)gm * HID_ + gn] = acc[m][n][r];
      }
}

// ---------------------------------------------------------------------------
extern "C" void kernel_launch(void* const* d_in, const int* in_sizes, int n_in,
                              void* d_out, int out_size, void* d_ws, size_t ws_size,
                              hipStream_t stream) {
  const int*   positions = (const int*)d_in[0];
  const float* hidden    = (const float*)d_in[1];
  const float* w_qkv     = (const float*)d_in[2];
  const float* w_o       = (const float*)d_in[3];
  float*       out       = (float*)d_out;

  const size_t qElems  = (size_t)B_ * S_ * NH_ * HD_;   // 16M
  const size_t kvElems = (size_t)B_ * S_ * NKV_ * HD_;  // 4M
  __bf16* qbuf = (__bf16*)d_ws;
  __bf16* kbuf = qbuf + qElems;
  __bf16* vbuf = kbuf + kvElems;
  __bf16* abuf = vbuf + kvElems;  // B*S*QSZ_ elems

  qkv_gemm_kernel<<<dim3(48, 32), 256, 0, stream>>>(hidden, w_qkv, qbuf, kbuf, vbuf);

  const size_t ropeTotal = (size_t)B_ * S_ * (NH_ + NKV_) * HALF_;
  rope_kernel<<<dim3((unsigned)((ropeTotal + 255) / 256)), 256, 0, stream>>>(
      positions, qbuf, kbuf);

  flash_attn_kernel<<<dim3(B_ * NH_, S_ / 128), 256, 0, stream>>>(qbuf, kbuf, vbuf,
                                                                  abuf);

  oproj_gemm_kernel<<<dim3(32, 32), 256, 0, stream>>>(abuf, w_o, out);
}
